// Self_Attn_Discriminator_7593502179388
// MI455X (gfx1250) — matrix operand order repaired and checked by reading, not compiled, not run on hardware
//
#include <hip/hip_runtime.h>

// ---------------------------------------------------------------------------
// Self-attention (SAGAN-style) for MI455X / gfx1250, wave32 + WMMA bf16.
//
// Pipeline (touches the 536MB attention region exactly once):
//   1) proj_kernel      : Q,K bf16 [B][N][32], V bf16 [B][C][N]  (workspace)
//   2) attn_stats_kernel: pass1 rowmax (1 v_max/elem), pass2 recompute-E
//                         sumexp (fma+exp2+add per elem); both via WMMA tiles
//   3) attn_out_kernel  : 8 waves/block, V chunks staged to LDS with
//                         global_load_async_to_lds_b128 (double buffered,
//                         ASYNCcnt), E recomputed via WMMA, P=exp(E-m)/l
//                         written non-temporally, P@V^T accumulated over all
//                         256 channels, fused residual epilogue.
// ---------------------------------------------------------------------------

typedef __attribute__((ext_vector_type(16))) __bf16 v16bf;
typedef __attribute__((ext_vector_type(8)))  __bf16 v8bf;
typedef __attribute__((ext_vector_type(8)))  float  v8f;

union V16 { v16bf v; v8bf h[2]; };

#define NB   8
#define NC   256
#define NCQ  32
#define NN   4096            // 64*64
#define L2E  1.4426950408889634f
#define VPAD 40              // padded LDS row stride (elems) -> conflict-free
#define VBUFE (NC * VPAD)    // one V staging buffer: 256 rows x 40 x 2B = 20KB

__device__ __forceinline__ __bf16 f2bf(float f) {
  unsigned u = __builtin_bit_cast(unsigned, f);
  unsigned r = u + 0x7fffu + ((u >> 16) & 1u);   // round-to-nearest-even
  unsigned short h = (unsigned short)(r >> 16);
  return __builtin_bit_cast(__bf16, h);
}

// ---------------------------------------------------------------------------
// Kernel 1: per-pixel linear projection (1x1 conv).
// grid = (N/64, OC/32, B), block = 256. Each thread: 1 pixel x 8 channels.
// layout==0 -> Y[(b*N+n)*OCtot + o]  (Q/K)   layout==1 -> Y[(b*C+o)*N + n] (V)
// ---------------------------------------------------------------------------
__global__ void proj_kernel(const float* __restrict__ x,
                            const float* __restrict__ W,
                            const float* __restrict__ bias,
                            __bf16* __restrict__ Y,
                            int layout, int OCtot) {
  __shared__ float Wl[32 * NC];
  const int tid = threadIdx.x;
  const int b  = blockIdx.z;
  const int o0 = blockIdx.y * 32;
  const int n0 = blockIdx.x * 64;

  for (int i = tid; i < 32 * NC; i += 256) Wl[i] = W[(size_t)o0 * NC + i];
  __syncthreads();

  const int px = tid & 63;
  const int og = tid >> 6;
  float acc[8];
#pragma unroll
  for (int j = 0; j < 8; ++j) acc[j] = 0.f;

  const float* xp = x + ((size_t)b * NC) * NN + n0 + px;
  for (int c = 0; c < NC; ++c) {
    float xv = xp[(size_t)c * NN];
#pragma unroll
    for (int j = 0; j < 8; ++j) acc[j] += Wl[(og * 8 + j) * NC + c] * xv;
  }
#pragma unroll
  for (int j = 0; j < 8; ++j) {
    int o = o0 + og * 8 + j;
    __bf16 hv = f2bf(acc[j] + bias[o]);
    if (layout == 0)
      Y[((size_t)(b * NN + n0 + px)) * OCtot + o] = hv;
    else
      Y[((size_t)(b * NC + o)) * NN + n0 + px] = hv;
  }
}

// --- fragment loaders (ISA 7.12.2 layouts) ---------------------------------
__device__ __forceinline__ v16bf load_a_q(const __bf16* Qb, int b, int n0, int lane) {
  const int row = n0 + (lane & 15);
  const int kb  = (lane >> 4) << 3;        // 0 or 8
  const __bf16* p = Qb + ((size_t)(b * NN + row)) * NCQ;
  V16 a;
  a.h[0] = *(const v8bf*)(p + kb);
  a.h[1] = *(const v8bf*)(p + kb + 16);
  return a.v;
}
__device__ __forceinline__ v16bf load_b_k(const __bf16* Kb, int b, int m0, int lane) {
  const int col = m0 + (lane & 15);
  const int kb  = (lane >> 4) << 3;        // 0 or 8
  const __bf16* p = Kb + ((size_t)(b * NN + col)) * NCQ;
  V16 r;
  r.h[0] = *(const v8bf*)(p + kb);
  r.h[1] = *(const v8bf*)(p + kb + 16);
  return r.v;
}

// ---------------------------------------------------------------------------
// Kernel 2: softmax row stats, two passes (max, then recompute-E sumexp).
// grid = (N/64, B), block = 128 (4 waves, 16 rows each).
// ---------------------------------------------------------------------------
__global__ void attn_stats_kernel(const __bf16* __restrict__ Qb,
                                  const __bf16* __restrict__ Kb,
                                  float2* __restrict__ stats) {
  const int lane = threadIdx.x & 31;
  const int wv   = threadIdx.x >> 5;
  const int b    = blockIdx.y;
  const int n0   = blockIdx.x * 64 + wv * 16;

  const v16bf afrag = load_a_q(Qb, b, n0, lane);
  v8f zf = {0.f, 0.f, 0.f, 0.f, 0.f, 0.f, 0.f, 0.f};

  // ---- pass 1: row max ----
  float m[8];
#pragma unroll
  for (int r = 0; r < 8; ++r) m[r] = -1e30f;
  for (int mt = 0; mt < NN / 16; ++mt) {
    v16bf bfrag = load_b_k(Kb, b, mt * 16, lane);
    v8f e = __builtin_amdgcn_wmma_f32_16x16x32_bf16(
        false, afrag, false, bfrag, (short)0, zf, false, false);
#pragma unroll
    for (int r = 0; r < 8; ++r) m[r] = fmaxf(m[r], e[r]);
  }
#pragma unroll
  for (int r = 0; r < 8; ++r)
#pragma unroll
    for (int off = 1; off < 16; off <<= 1)
      m[r] = fmaxf(m[r], __shfl_xor(m[r], off, 32));

  float mlog[8], l[8];
#pragma unroll
  for (int r = 0; r < 8; ++r) { mlog[r] = m[r] * L2E; l[r] = 0.f; }

  // ---- pass 2: recompute E, accumulate sum of exp2(e*log2e - m*log2e) ----
  for (int mt = 0; mt < NN / 16; ++mt) {
    v16bf bfrag = load_b_k(Kb, b, mt * 16, lane);
    v8f e = __builtin_amdgcn_wmma_f32_16x16x32_bf16(
        false, afrag, false, bfrag, (short)0, zf, false, false);
#pragma unroll
    for (int r = 0; r < 8; ++r)
      l[r] += exp2f(fmaf(e[r], L2E, -mlog[r]));
  }
#pragma unroll
  for (int r = 0; r < 8; ++r)
#pragma unroll
    for (int off = 1; off < 16; off <<= 1)
      l[r] += __shfl_xor(l[r], off, 32);

  if ((lane & 15) == 0) {
    int nb = n0 + ((lane >> 4) << 3);
#pragma unroll
    for (int r = 0; r < 8; ++r)
      stats[(size_t)b * NN + nb + r] = make_float2(m[r], l[r]);
  }
}

// --- async stage one 32-m V chunk (256 ch x 32 m bf16, padded rows) --------
__device__ __forceinline__ void async_v_chunk(const __bf16* __restrict__ Vb,
                                              int b, int m0,
                                              __bf16* ldsbuf, int t) {
  const int c0   = t >> 2;          // 0..63
  const int moff = (t & 3) * 8;     // 0,8,16,24 elems (16B granules)
#pragma unroll
  for (int i = 0; i < 4; ++i) {     // 4 x b128 per thread = 16KB per block
    int c = c0 + i * 64;
    const __bf16* src = Vb + ((size_t)(b * NC + c)) * NN + m0 + moff;
    unsigned dst = (unsigned)(uintptr_t)(ldsbuf + c * VPAD + moff);
    asm volatile("global_load_async_to_lds_b128 %0, %1, off"
                 :: "v"(dst), "v"((unsigned long long)(uintptr_t)src)
                 : "memory");
  }
}

// ---------------------------------------------------------------------------
// Kernel 3: recompute E, write normalized attention (only N^2 write, NT),
// accumulate out = P @ V^T with V chunks double-buffered in LDS via async DMA.
// grid = (N/128, B), block = 256 (8 waves, 16 rows each).
// ---------------------------------------------------------------------------
__global__ void attn_out_kernel(const __bf16* __restrict__ Qb,
                                const __bf16* __restrict__ Kb,
                                const __bf16* __restrict__ Vb,
                                const float2* __restrict__ stats,
                                const float* __restrict__ x,
                                const float* __restrict__ gammap,
                                float* __restrict__ attn,
                                float* __restrict__ out) {
  __shared__ __bf16 ldsV[2 * VBUFE];          // 2 x 20KB double buffer
  __shared__ __bf16 ldsP[8 * 16 * VPAD];      // per-wave padded P tiles (10KB)

  const int tid  = threadIdx.x;
  const int lane = tid & 31;
  const int wv   = tid >> 5;
  const int b    = blockIdx.y;
  const int n0   = blockIdx.x * 128 + wv * 16;

  const v16bf afrag = load_a_q(Qb, b, n0, lane);
  v8f zf = {0.f, 0.f, 0.f, 0.f, 0.f, 0.f, 0.f, 0.f};

  const int half8 = (lane >> 4) << 3;         // 0 / 8
  const int col16 = lane & 15;
  float mlog[8], rinv[8];
#pragma unroll
  for (int r = 0; r < 8; ++r) {
    float2 s = stats[(size_t)b * NN + n0 + half8 + r];
    mlog[r] = s.x * L2E;
    rinv[r] = 1.f / s.y;
  }

  v8f acc[16];
#pragma unroll
  for (int ct = 0; ct < 16; ++ct)
#pragma unroll
    for (int i = 0; i < 8; ++i) acc[ct][i] = 0.f;

  __bf16* myP  = ldsP + wv * 16 * VPAD;
  float* attnB = attn + (size_t)b * NN * NN;

  async_v_chunk(Vb, b, 0, ldsV, tid);          // prologue: chunk 0 -> buf 0

  for (int kc = 0; kc < NN / 32; ++kc) {
    const int buf = kc & 1;
    __syncthreads();                           // buf^1 free for overwrite
    if (kc + 1 < NN / 32) {
      async_v_chunk(Vb, b, (kc + 1) * 32, ldsV + (buf ^ 1) * VBUFE, tid);
      asm volatile("s_wait_asynccnt 4" ::: "memory");   // chunk kc landed
    } else {
      asm volatile("s_wait_asynccnt 0" ::: "memory");
    }
    __syncthreads();                           // all threads' chunk kc visible

#pragma unroll
    for (int s = 0; s < 2; ++s) {
      const int mt = kc * 2 + s;
      // ---- energy tile (1 WMMA, K = Cq = 32) ----
      v16bf bfrag = load_b_k(Kb, b, mt * 16, lane);
      v8f e = __builtin_amdgcn_wmma_f32_16x16x32_bf16(
          false, afrag, false, bfrag, (short)0, zf, false, false);

      // ---- P = exp(E - m)/l : NT-store attention, stage bf16 P in LDS ----
      const int cb = s << 4;
      size_t arow = ((size_t)(n0 + half8)) * NN + (size_t)mt * 16 + col16;
#pragma unroll
      for (int r = 0; r < 8; ++r) {
        float p = exp2f(fmaf(e[r], L2E, -mlog[r])) * rinv[r];
        __builtin_nontemporal_store(p, &attnB[arow + (size_t)r * NN]);
        myP[(half8 + r) * VPAD + cb + col16] = f2bf(p);
      }
    }

    // ---- out += P(16x32) @ V^T(32x16) over all 16 channel tiles ----
    asm volatile("s_wait_dscnt 0" ::: "memory");   // wave-private P RAW
    {
      const int prow = lane & 15;
      const int pkb  = (lane >> 4) << 3;
      V16 pf;
      pf.h[0] = *(const v8bf*)(myP + prow * VPAD + pkb);
      pf.h[1] = *(const v8bf*)(myP + prow * VPAD + pkb + 16);

      const __bf16* vb0 = ldsV + buf * VBUFE + ((lane >> 4) << 3);
#pragma unroll
      for (int ct = 0; ct < 16; ++ct) {
        const __bf16* vp = vb0 + (ct * 16 + col16) * VPAD;
        V16 bf;
        bf.h[0] = *(const v8bf*)(vp);
        bf.h[1] = *(const v8bf*)(vp + 16);
        acc[ct] = __builtin_amdgcn_wmma_f32_16x16x32_bf16(
            false, pf.v, false, bf.v, (short)0, acc[ct], false, false);
      }
    }
  }

  // ---- epilogue: out = 0.5*gamma*acc + x ----
  const float g = 0.5f * gammap[0];
#pragma unroll
  for (int ct = 0; ct < 16; ++ct) {
    int c = ct * 16 + col16;
    size_t base = ((size_t)(b * NC + c)) * NN + n0 + half8;
#pragma unroll
    for (int r = 0; r < 8; ++r)
      out[base + r] = g * acc[ct][r] + x[base + r];
  }
}

// ---------------------------------------------------------------------------
extern "C" void kernel_launch(void* const* d_in, const int* in_sizes, int n_in,
                              void* d_out, int out_size, void* d_ws, size_t ws_size,
                              hipStream_t stream) {
  (void)in_sizes; (void)n_in; (void)out_size; (void)ws_size;
  const float* x     = (const float*)d_in[0];
  const float* Wq    = (const float*)d_in[1];
  const float* bq    = (const float*)d_in[2];
  const float* Wk    = (const float*)d_in[3];
  const float* bk    = (const float*)d_in[4];
  const float* Wv    = (const float*)d_in[5];
  const float* bv    = (const float*)d_in[6];
  const float* gamma = (const float*)d_in[7];

  float* out  = (float*)d_out;                       // B*C*N fp32
  float* attn = out + (size_t)NB * NC * NN;          // B*N*N fp32

  __bf16* qws = (__bf16*)d_ws;                       // 2MB
  __bf16* kws = qws + (size_t)NB * NN * NCQ;         // 2MB
  __bf16* vws = kws + (size_t)NB * NN * NCQ;         // 16MB
  float2* st  = (float2*)(vws + (size_t)NB * NC * NN);  // 256KB

  proj_kernel<<<dim3(NN / 64, 1, NB), 256, 0, stream>>>(x, Wq, bq, qws, 0, NCQ);
  proj_kernel<<<dim3(NN / 64, 1, NB), 256, 0, stream>>>(x, Wk, bk, kws, 0, NCQ);
  proj_kernel<<<dim3(NN / 64, 8, NB), 256, 0, stream>>>(x, Wv, bv, vws, 1, NC);

  attn_stats_kernel<<<dim3(NN / 64, NB), 128, 0, stream>>>(qws, kws, st);
  attn_out_kernel<<<dim3(NN / 128, NB), 256, 0, stream>>>(qws, kws, vws, st,
                                                          x, gamma, attn, out);
}
